// G2_68350109548985
// MI455X (gfx1250) — compile-verified, matrix operand order checked by reading, not run on hardware
//
#include <hip/hip_runtime.h>
#include <hip/hip_bf16.h>
#include <cstdint>
#include <cstddef>

typedef __attribute__((ext_vector_type(16))) _Float16 v16h;
typedef __attribute__((ext_vector_type(8)))  _Float16 v8h;
typedef __attribute__((ext_vector_type(2)))  _Float16 v2h;
typedef __attribute__((ext_vector_type(8)))  float    v8f;

#define BATCH 8
#define NNODE 2048
#define DIN   256
#define DOUT  256

// Convert four float4 (16 fp32) into a 16xf16 WMMA fragment (K-order already
// arranged by the caller's address math: halves 0..7 = first 32B segment,
// halves 8..15 = second 32B segment).
__device__ __forceinline__ v16h pack_f32x16(const float4 a0, const float4 a1,
                                            const float4 a2, const float4 a3) {
  v16h r;
  r[0]  = (_Float16)a0.x; r[1]  = (_Float16)a0.y; r[2]  = (_Float16)a0.z; r[3]  = (_Float16)a0.w;
  r[4]  = (_Float16)a1.x; r[5]  = (_Float16)a1.y; r[6]  = (_Float16)a1.z; r[7]  = (_Float16)a1.w;
  r[8]  = (_Float16)a2.x; r[9]  = (_Float16)a2.y; r[10] = (_Float16)a2.z; r[11] = (_Float16)a2.w;
  r[12] = (_Float16)a3.x; r[13] = (_Float16)a3.y; r[14] = (_Float16)a3.z; r[15] = (_Float16)a3.w;
  return r;
}

// ---------------------------------------------------------------------------
// Stage 1: X = relu(features @ W^T + b)   (f16 WMMA, fp32 accum)
//   - writes X transposed:  XT[b][e][n]  (f16)  -> stage-2 B operand friendly
//   - writes sq[b][n] = sum_e X[b,n,e]^2 (f32)
// Block: 256 threads (8 wave32), 16 n-rows x 256 e-cols. Wave w owns e-tiles
// 2w and 2w+1. K loop over DIN in steps of 32, fully unrolled (16 WMMAs).
// ---------------------------------------------------------------------------
__global__ __launch_bounds__(256) void g2_stage1_gemm_relu(
    const float* __restrict__ feat, const float* __restrict__ W,
    const float* __restrict__ bias, _Float16* __restrict__ XT,
    float* __restrict__ sqout)
{
  __shared__ float sq_l[16];
  const int tid  = threadIdx.x;
  const int wave = tid >> 5;     // wave32
  const int lane = tid & 31;
  const int lrow = lane & 15;    // M (or N) index within 16
  const int g    = lane >> 4;    // K half-group select
  const int b    = blockIdx.x >> 7;
  const int n0   = (blockIdx.x & 127) << 4;

  if (tid < 16) sq_l[tid] = 0.0f;
  __syncthreads();

  const int n  = n0 + lrow;                 // A-matrix row (M)
  const int e0 = 32 * wave + lrow;          // B/D column (N), tile 0
  const int e1 = e0 + 16;                   // tile 1
  const float* frow  = feat + ((size_t)b * NNODE + n) * DIN + 8 * g;
  const float* wrow0 = W + (size_t)e0 * DIN + 8 * g;
  const float* wrow1 = W + (size_t)e1 * DIN + 8 * g;

  v8f acc0 = {}; v8f acc1 = {};
#pragma unroll
  for (int d0 = 0; d0 < DIN; d0 += 32) {
    v16h af = pack_f32x16(*(const float4*)(frow + d0),
                          *(const float4*)(frow + d0 + 4),
                          *(const float4*)(frow + d0 + 16),
                          *(const float4*)(frow + d0 + 20));
    v16h bf0 = pack_f32x16(*(const float4*)(wrow0 + d0),
                           *(const float4*)(wrow0 + d0 + 4),
                           *(const float4*)(wrow0 + d0 + 16),
                           *(const float4*)(wrow0 + d0 + 20));
    v16h bf1 = pack_f32x16(*(const float4*)(wrow1 + d0),
                           *(const float4*)(wrow1 + d0 + 4),
                           *(const float4*)(wrow1 + d0 + 16),
                           *(const float4*)(wrow1 + d0 + 20));
    acc0 = __builtin_amdgcn_wmma_f32_16x16x32_f16(false, af, false, bf0,
                                                  (short)0, acc0, false, false);
    acc1 = __builtin_amdgcn_wmma_f32_16x16x32_f16(false, af, false, bf1,
                                                  (short)0, acc1, false, false);
  }

  const float bz0 = bias[e0];
  const float bz1 = bias[e1];
  v8h p0, p1;
  float sqc[8];
#pragma unroll
  for (int r = 0; r < 8; ++r) {
    float x0 = fmaxf(acc0[r] + bz0, 0.0f);
    float x1 = fmaxf(acc1[r] + bz1, 0.0f);
    p0[r] = (_Float16)x0;
    p1[r] = (_Float16)x1;
    sqc[r] = x0 * x0 + x1 * x1;      // this lane covers cols e0,e1 of row n0+8g+r
  }
  // D tile row M = r + 8g  ->  node index n0 + 8g + r : contiguous 8 halves.
  *(v8h*)(XT + ((size_t)b * DOUT + e0) * NNODE + n0 + 8 * g) = p0;
  *(v8h*)(XT + ((size_t)b * DOUT + e1) * NNODE + n0 + 8 * g) = p1;
#pragma unroll
  for (int r = 0; r < 8; ++r) atomicAdd(&sq_l[8 * g + r], sqc[r]);
  __syncthreads();
  if (tid < 16) sqout[(size_t)b * NNODE + n0 + tid] = sq_l[tid];
}

// ---------------------------------------------------------------------------
// Stage 2: per (b, 16-row tile i):
//   S_i = sum_j adj_ij * X_j  (f16 WMMA over K=j, fp32 accum)
//   adj tile built ONCE per block into a double-buffered LDS tile
//   (coalesced float2 reads of support; cmp/select/cvt done by 1 thread per
//   2 elements instead of once per wave). count_i / nbr_sq_i accumulate in
//   the builder threads' registers and commit once via ds_add_f32.
//   tau_i = tanh(max(count*sq_i + nbr_sq - 2*X_i.S_i, 0) / max(count,1))
// ---------------------------------------------------------------------------
__global__ __launch_bounds__(256) void g2_stage2_graph_tau(
    const float* __restrict__ support, const float* __restrict__ maskp,
    const _Float16* __restrict__ XT, const float* __restrict__ sqin,
    float* __restrict__ out)
{
  __shared__ __align__(16) _Float16 adj_l[2][16][32];  // double-buffered A tile
  __shared__ float S_l[16][260];                       // 16 x 256 (+pad)
  __shared__ float cnt_l[16];
  __shared__ float nbr_l[16];
  __shared__ float dot_l[16];

  const int tid  = threadIdx.x;
  const int wave = tid >> 5;
  const int lane = tid & 31;
  const int lrow = lane & 15;
  const int g    = lane >> 4;
  const int b    = blockIdx.x >> 7;
  const int i0   = (blockIdx.x & 127) << 4;

  if (tid < 16) { cnt_l[tid] = 0.0f; nbr_l[tid] = 0.0f; dot_l[tid] = 0.0f; }

  // ---- builder mapping: thread -> (row r, col pair jj) of the 16x32 tile
  const int br = tid >> 4;          // 0..15 : adj row (node i0+br)
  const int bj = (tid & 15) * 2;    // 0,2,..,30 : col pair within tile
  const float mib = maskp[(size_t)b * NNODE + i0 + br];
  const float* srow_t = support + ((size_t)b * NNODE + i0 + br) * NNODE + bj;
  const float* mrow_t = maskp + (size_t)b * NNODE + bj;
  const float* qrow_t = sqin  + (size_t)b * NNODE + bj;
  float cnt_t = 0.0f, nbr_t = 0.0f;

  // ---- per-wave GEMM operand pointers
  const int c0 = 32 * wave + lrow;                       // S column, tile 0
  const _Float16* xr0 = XT + ((size_t)b * DOUT + c0)      * NNODE + 8 * g;
  const _Float16* xr1 = XT + ((size_t)b * DOUT + c0 + 16) * NNODE + 8 * g;

  v8f acc0 = {}; v8f acc1 = {};

  // build first tile (j0 = 0)
  {
    const float2 s = *(const float2*)(srow_t);
    const float2 m = *(const float2*)(mrow_t);
    const float2 q = *(const float2*)(qrow_t);
    const float a0 = (s.x > 0.0f) ? (mib * m.x) : 0.0f;
    const float a1 = (s.y > 0.0f) ? (mib * m.y) : 0.0f;
    cnt_t += a0 + a1;
    nbr_t += a0 * q.x + a1 * q.y;
    v2h pa; pa[0] = (_Float16)a0; pa[1] = (_Float16)a1;
    *(v2h*)&adj_l[0][br][bj] = pa;
  }
  __syncthreads();

  for (int j0 = 0; j0 < NNODE; j0 += 32) {
    const int buf = (j0 >> 5) & 1;

    // prefetch the support stream a few tiles ahead (-> global_prefetch_b8)
    if (j0 + 256 < NNODE) __builtin_prefetch(srow_t + j0 + 256, 0, 1);

    // build next tile into the other buffer (no barrier needed: previous
    // barrier guarantees all reads of buf^1 from iter j0-32 are complete)
    if (j0 + 32 < NNODE) {
      const float2 s = *(const float2*)(srow_t + j0 + 32);
      const float2 m = *(const float2*)(mrow_t + j0 + 32);
      const float2 q = *(const float2*)(qrow_t + j0 + 32);
      const float a0 = (s.x > 0.0f) ? (mib * m.x) : 0.0f;
      const float a1 = (s.y > 0.0f) ? (mib * m.y) : 0.0f;
      cnt_t += a0 + a1;
      nbr_t += a0 * q.x + a1 * q.y;
      v2h pa; pa[0] = (_Float16)a0; pa[1] = (_Float16)a1;
      *(v2h*)&adj_l[buf ^ 1][br][bj] = pa;
    }

    // A fragment from LDS: two ds_load_b128 per lane (broadcast to 8 waves)
    const v8h alo = *(const v8h*)&adj_l[buf][lrow][8 * g];       // h = 0..7
    const v8h ahi = *(const v8h*)&adj_l[buf][lrow][16 + 8 * g];  // h = 8..15
    // B operand: X_T rows (feature d fixed per lane), contiguous f16 chunks.
    const v8h lo0 = *(const v8h*)(xr0 + j0);
    const v8h hi0 = *(const v8h*)(xr0 + j0 + 16);
    const v8h lo1 = *(const v8h*)(xr1 + j0);
    const v8h hi1 = *(const v8h*)(xr1 + j0 + 16);

    v16h af, bf0, bf1;
#pragma unroll
    for (int t = 0; t < 8; ++t) {
      af[t]  = alo[t]; af[8 + t]  = ahi[t];
      bf0[t] = lo0[t]; bf0[8 + t] = hi0[t];
      bf1[t] = lo1[t]; bf1[8 + t] = hi1[t];
    }

    acc0 = __builtin_amdgcn_wmma_f32_16x16x32_f16(false, af, false, bf0,
                                                  (short)0, acc0, false, false);
    acc1 = __builtin_amdgcn_wmma_f32_16x16x32_f16(false, af, false, bf1,
                                                  (short)0, acc1, false, false);

    __syncthreads();   // protect buf from overwrite; publish buf^1
  }

  // Spill S tile to LDS (row M = r + 8g, col = c0 / c0+16).
#pragma unroll
  for (int r = 0; r < 8; ++r) {
    S_l[8 * g + r][c0]      = acc0[r];
    S_l[8 * g + r][c0 + 16] = acc1[r];
  }
  // commit count / nbr_sq partials (16 builder threads per row)
  atomicAdd(&cnt_l[br], cnt_t);
  atomicAdd(&nbr_l[br], nbr_t);
  __syncthreads();

  // dot_i = X[i,:] . S[i,:]  (16 threads per row, 16 cols each)
  const int row   = tid & 15;
  const int chunk = tid >> 4;
  float part = 0.0f;
#pragma unroll
  for (int dd = 0; dd < 16; ++dd) {
    const int d = chunk * 16 + dd;
    const float x = (float)XT[((size_t)b * DOUT + d) * NNODE + i0 + row];
    part += x * S_l[row][d];
  }
  atomicAdd(&dot_l[row], part);
  __syncthreads();

  if (tid < 16) {
    const float c    = cnt_l[tid];
    const float sqi  = sqin[(size_t)b * NNODE + i0 + tid];
    const float diff = fmaxf(c * sqi + nbr_l[tid] - 2.0f * dot_l[tid], 0.0f);
    out[(size_t)b * NNODE + i0 + tid] = tanhf(diff / fmaxf(c, 1.0f));
  }
}

extern "C" void kernel_launch(void* const* d_in, const int* in_sizes, int n_in,
                              void* d_out, int out_size, void* d_ws, size_t ws_size,
                              hipStream_t stream) {
  const float* feat    = (const float*)d_in[0];   // [B,N,D]
  const float* support = (const float*)d_in[1];   // [B,N,N]
  const float* maskp   = (const float*)d_in[2];   // [B,N,1]
  const float* W       = (const float*)d_in[3];   // [D2,D]
  const float* bias    = (const float*)d_in[4];   // [D2]
  float* out = (float*)d_out;                     // [B,N]

  // Workspace: X transposed (f16)  +  per-node squared norms (f32)
  _Float16* XT = (_Float16*)d_ws;
  float* sq = (float*)((char*)d_ws + (size_t)BATCH * DOUT * NNODE * sizeof(_Float16));

  dim3 grid(BATCH * (NNODE / 16));
  g2_stage1_gemm_relu<<<grid, dim3(256), 0, stream>>>(feat, W, bias, XT, sq);
  g2_stage2_graph_tau<<<grid, dim3(256), 0, stream>>>(support, maskp, XT, sq, out);
}
